// OuroLayer_66511863545985
// MI455X (gfx1250) — compile-verified, matrix operand order checked by reading, not run on hardware
//
#include <hip/hip_runtime.h>
#include <math.h>

// ---------------------------------------------------------------------------
// OuroLayer forward for MI455X (gfx1250, wave32, WMMA).
// GEMMs: v_wmma_f32_16x16x32_bf16, 128x128 block tile, K-step 32,
// double-buffered LDS (ping-pong) with register-staged global loads so the
// loads for tile k+1 overlap the WMMAs of tile k. 8 waves, each owns a
// 32x64 patch (2x4 fragments -> 8 WMMAs per K-step).
// ---------------------------------------------------------------------------

typedef __bf16 bf16;
typedef __attribute__((ext_vector_type(16))) bf16  v16bf;
typedef __attribute__((ext_vector_type(8)))  bf16  v8bf;
typedef __attribute__((ext_vector_type(8)))  float v8f;
typedef __attribute__((ext_vector_type(4)))  float v4f;

#define BM 128
#define BN 128
#define BK 32
#define BKP 40   // padded LDS row (80B): 16B-aligned fragment loads, conflict-free

__device__ __forceinline__ float wred(float v) {
#pragma unroll
  for (int m = 16; m > 0; m >>= 1) v += __shfl_xor(v, m, 32);
  return v;
}

// Stage one K-tile of A (BMxBK) and B (BNxBK) into registers (pure loads).
__device__ __forceinline__ void load_tile(const float* __restrict__ Ab,
                                          const float* __restrict__ Bb,
                                          int m0, int n0, int N, int K, int k0,
                                          int transB, int tid,
                                          v4f (&ra)[4], v4f (&rb)[4]) {
#pragma unroll
  for (int it = 0; it < 4; ++it) {          // A: 1024 float4 groups
    int grp = tid + it * 256;
    int m = grp >> 3, kk = (grp & 7) << 2;  // 8 groups per 32-wide row
    ra[it] = *(const v4f*)&Ab[(long)(m0 + m) * K + k0 + kk];
  }
  if (transB) {
#pragma unroll
    for (int it = 0; it < 4; ++it) {        // B as (N x K): K contiguous
      int grp = tid + it * 256;
      int n = grp >> 3, kk = (grp & 7) << 2;
      rb[it] = *(const v4f*)&Bb[(long)(n0 + n) * K + k0 + kk];
    }
  } else {
#pragma unroll
    for (int it = 0; it < 4; ++it) {        // B as (K x N): N contiguous
      int grp = tid + it * 256;
      int kk = grp >> 5, n = (grp & 31) << 2;  // 32 groups per 128-wide k-row
      rb[it] = *(const v4f*)&Bb[(long)(k0 + kk) * N + n0 + n];
    }
  }
}

// Convert staged registers to bf16 and store into the LDS ping-pong buffer.
__device__ __forceinline__ void store_tile(bf16 (*__restrict__ As)[BKP],
                                           bf16 (*__restrict__ Bs)[BKP],
                                           int transB, int tid,
                                           const v4f (&ra)[4], const v4f (&rb)[4]) {
#pragma unroll
  for (int it = 0; it < 4; ++it) {
    int grp = tid + it * 256;
    int m = grp >> 3, kk = (grp & 7) << 2;
    As[m][kk + 0] = (bf16)ra[it].x;  As[m][kk + 1] = (bf16)ra[it].y;
    As[m][kk + 2] = (bf16)ra[it].z;  As[m][kk + 3] = (bf16)ra[it].w;
  }
  if (transB) {
#pragma unroll
    for (int it = 0; it < 4; ++it) {
      int grp = tid + it * 256;
      int n = grp >> 3, kk = (grp & 7) << 2;
      Bs[n][kk + 0] = (bf16)rb[it].x;  Bs[n][kk + 1] = (bf16)rb[it].y;
      Bs[n][kk + 2] = (bf16)rb[it].z;  Bs[n][kk + 3] = (bf16)rb[it].w;
    }
  } else {
#pragma unroll
    for (int it = 0; it < 4; ++it) {
      int grp = tid + it * 256;
      int kk = grp >> 5, n = (grp & 31) << 2;
      Bs[n + 0][kk] = (bf16)rb[it].x;  Bs[n + 1][kk] = (bf16)rb[it].y;
      Bs[n + 2][kk] = (bf16)rb[it].z;  Bs[n + 3][kk] = (bf16)rb[it].w;
    }
  }
}

// ---------------------------------------------------------------------------
// Generic WMMA GEMM:  C[bz] = epilogue( alpha * A[bz] @ B[bz] )
//   act: 0 none, 1 silu, 2 sigmoid.  addsrc: C += beta*addsrc (post-act).
//   transB: B given as (N x K) row-major (q @ k^T).
//   causal_mask: zero cols > row (scores);  causal_k: K-loop only to row max.
// ---------------------------------------------------------------------------
__global__ __launch_bounds__(256)
void wmma_gemm_kernel(const float* __restrict__ A, const float* __restrict__ Bm,
                      const float* __restrict__ bias, const float* __restrict__ addsrc,
                      float* __restrict__ C, int N, int K,
                      long sA, long sB, long sC, long sAdd,
                      int transB, int act, float alpha, float beta,
                      int causal_mask, int causal_k) {
  __shared__ bf16 As[2][BM][BKP];
  __shared__ bf16 Bs[2][BN][BKP];

  const int tid  = threadIdx.x;
  const int lane = tid & 31;
  const int wav  = tid >> 5;
  const int waveM = wav & 3;        // 4 waves along M (32 rows each)
  const int waveN = wav >> 2;       // 2 waves along N (64 cols each)
  const int half  = lane >> 4;
  const int lrow  = lane & 15;

  const int m0 = blockIdx.y * BM;
  const int n0 = blockIdx.x * BN;
  const int bz = blockIdx.z;

  const float* Ab = A  + (long)bz * sA;
  const float* Bb = Bm + (long)bz * sB;
  float*       Cb = C  + (long)bz * sC;
  const float* Addb = addsrc ? addsrc + (long)bz * sAdd : nullptr;

  const v8f vz = {0.f, 0.f, 0.f, 0.f, 0.f, 0.f, 0.f, 0.f};
  v8f acc[2][4] = {{vz, vz, vz, vz}, {vz, vz, vz, vz}};

  int kmax = K;
  if (causal_k) { int lim = m0 + BM; kmax = lim < K ? lim : K; }
  if (causal_mask && (m0 + BM - 1) < n0) kmax = 0;   // fully-masked block

  v4f ra[4], rb[4];
  if (kmax > 0) {                                    // prologue: tile 0
    load_tile(Ab, Bb, m0, n0, N, K, 0, transB, tid, ra, rb);
    store_tile(As[0], Bs[0], transB, tid, ra, rb);
    __syncthreads();
  }

  for (int k0 = 0; k0 < kmax; k0 += BK) {
    const int cur = (k0 >> 5) & 1;
    const bool more = (k0 + BK) < kmax;

    // Issue next tile's global loads before compute (latency hidden by WMMA).
    if (more) load_tile(Ab, Bb, m0, n0, N, K, k0 + BK, transB, tid, ra, rb);
    if (k0 + 2 * BK < kmax) {                        // warm GL2 for K+2 tile
      int m = tid >> 3, kk = (tid & 7) << 2;
      __builtin_prefetch(&Ab[(long)(m0 + m) * K + k0 + 2 * BK + kk], 0, 1);
      if (transB)
        __builtin_prefetch(&Bb[(long)(n0 + m) * K + k0 + 2 * BK + kk], 0, 1);
      else
        __builtin_prefetch(&Bb[(long)(k0 + 2 * BK + (tid >> 5)) * N + n0 + ((tid & 31) << 2)], 0, 1);
    }

    // Fragments per ISA 16-bit layout:
    // lanes 0-15: K {0..7, 16..23};  lanes 16-31: K {8..15, 24..31}
    const bf16 (*Asc)[BKP] = As[cur];
    const bf16 (*Bsc)[BKP] = Bs[cur];
    v16bf af[2], bfr[4];
#pragma unroll
    for (int i = 0; i < 2; ++i) {
      int m = waveM * 32 + i * 16 + lrow;
      v8bf lo = *(const v8bf*)&Asc[m][half * 8];
      v8bf hi = *(const v8bf*)&Asc[m][16 + half * 8];
      af[i] = __builtin_shufflevector(lo, hi, 0, 1, 2, 3, 4, 5, 6, 7,
                                      8, 9, 10, 11, 12, 13, 14, 15);
    }
#pragma unroll
    for (int j = 0; j < 4; ++j) {
      int n = waveN * 64 + j * 16 + lrow;
      v8bf lo = *(const v8bf*)&Bsc[n][half * 8];
      v8bf hi = *(const v8bf*)&Bsc[n][16 + half * 8];
      bfr[j] = __builtin_shufflevector(lo, hi, 0, 1, 2, 3, 4, 5, 6, 7,
                                       8, 9, 10, 11, 12, 13, 14, 15);
    }
#pragma unroll
    for (int i = 0; i < 2; ++i)
#pragma unroll
      for (int j = 0; j < 4; ++j)
        acc[i][j] = __builtin_amdgcn_wmma_f32_16x16x32_bf16(
            false, af[i], false, bfr[j], (short)0, acc[i][j], false, false);

    if (more) store_tile(As[cur ^ 1], Bs[cur ^ 1], transB, tid, ra, rb);
    __syncthreads();
  }

  // Epilogue: C/D layout -> row = r + 8*half, col = lane&15.
#pragma unroll
  for (int i = 0; i < 2; ++i)
#pragma unroll
    for (int j = 0; j < 4; ++j)
#pragma unroll
      for (int r = 0; r < 8; ++r) {
        int gm = m0 + waveM * 32 + i * 16 + r + 8 * half;
        int gn = n0 + waveN * 64 + j * 16 + lrow;
        float t = acc[i][j][r] * alpha;
        if (bias) t += bias[gn];
        if (act == 1)      t = t / (1.f + __expf(-t));     // silu
        else if (act == 2) t = 1.f / (1.f + __expf(-t));   // sigmoid
        if (Addb) t += beta * Addb[(long)gm * N + gn];
        if (causal_mask && gn > gm) t = 0.f;
        Cb[(long)gm * N + gn] = t;
      }
}

// ---------------------------------------------------------------------------
// LayerNorm over last dim N (block per row): out = res + LN(in + pre)
// ---------------------------------------------------------------------------
__global__ __launch_bounds__(256)
void ln_kernel(const float* __restrict__ in, const float* __restrict__ pre,
               const float* __restrict__ res, const float* __restrict__ g,
               const float* __restrict__ b, float* __restrict__ out, int N) {
  __shared__ float sh1[8], sh2[8];
  __shared__ float s_mean, s_rstd;
  const long off = (long)blockIdx.x * N;
  const int tid = threadIdx.x;
  float s = 0.f, s2 = 0.f;
  for (int j = tid; j < N; j += 256) {
    float v = in[off + j];
    if (pre) v += pre[off + j];
    s += v; s2 += v * v;
  }
  s = wred(s); s2 = wred(s2);
  if ((tid & 31) == 0) { sh1[tid >> 5] = s; sh2[tid >> 5] = s2; }
  __syncthreads();
  if (tid == 0) {
    float S = 0.f, S2 = 0.f;
#pragma unroll
    for (int i = 0; i < 8; ++i) { S += sh1[i]; S2 += sh2[i]; }
    float mean = S / N;
    s_mean = mean;
    s_rstd = rsqrtf(S2 / N - mean * mean + 1e-5f);
  }
  __syncthreads();
  const float mean = s_mean, rstd = s_rstd;
  for (int j = tid; j < N; j += 256) {
    float v = in[off + j];
    if (pre) v += pre[off + j];
    float o = (v - mean) * rstd * g[j] + b[j];
    if (res) o += res[off + j];
    out[off + j] = o;
  }
}

// ---------------------------------------------------------------------------
// Sequential cumsum over L for each (b, e) column (coalesced over e).
// ---------------------------------------------------------------------------
__global__ void cumsum_kernel(const float* __restrict__ in, float* __restrict__ out,
                              int Bc, int Lc, int Ec) {
  int col = blockIdx.x * blockDim.x + threadIdx.x;
  if (col >= Bc * Ec) return;
  int b = col / Ec, e = col - b * Ec;
  const float* ip = in + (long)b * Lc * Ec + e;
  float* op = out + (long)b * Lc * Ec + e;
  float acc = 0.f;
  for (int l = 0; l < Lc; ++l) {
    acc += ip[(long)l * Ec];
    op[(long)l * Ec] = acc;
  }
}

// ---------------------------------------------------------------------------
// Per-head gate: one wave32 per (b,l,h) row of D=64.
// gated = x_h * sigmoid( dot(LN(x_h), LN(xc_h)) / sqrt(D) )
// ---------------------------------------------------------------------------
__global__ __launch_bounds__(256)
void align_gate_kernel(const float* __restrict__ xmlp, const float* __restrict__ xc,
                       const float* __restrict__ gH, const float* __restrict__ bH,
                       float* __restrict__ gated, int nrows) {
  int gt = blockIdx.x * 256 + threadIdx.x;
  int row = gt >> 5, lane = gt & 31;
  if (row >= nrows) return;
  const float* xr = xmlp + (long)row * 64;
  const float* cr = xc + (long)row * 64;
  float x0 = xr[lane], x1 = xr[lane + 32];
  float c0 = cr[lane], c1 = cr[lane + 32];
  const float inv = 1.f / 64.f;
  float mx = wred(x0 + x1) * inv;
  float mc = wred(c0 + c1) * inv;
  float dx0 = x0 - mx, dx1 = x1 - mx, dc0 = c0 - mc, dc1 = c1 - mc;
  float rx = rsqrtf(wred(dx0 * dx0 + dx1 * dx1) * inv + 1e-5f);
  float rc = rsqrtf(wred(dc0 * dc0 + dc1 * dc1) * inv + 1e-5f);
  float g0 = gH[lane], g1 = gH[lane + 32], b0 = bH[lane], b1 = bH[lane + 32];
  float nx0 = dx0 * rx * g0 + b0, nx1 = dx1 * rx * g1 + b1;
  float nc0 = dc0 * rc * g0 + b0, nc1 = dc1 * rc * g1 + b1;
  float dot = wred(nx0 * nc0 + nx1 * nc1) * 0.125f;  // / sqrt(64)
  float sig = 1.f / (1.f + __expf(-dot));
  float* orow = gated + (long)row * 64;
  orow[lane] = x0 * sig;
  orow[lane + 32] = x1 * sig;
}

// ---------------------------------------------------------------------------
// k <- k / max(||k||, 1e-5) per row of length N.
// ---------------------------------------------------------------------------
__global__ __launch_bounds__(256)
void knorm_kernel(float* __restrict__ k, int N) {
  __shared__ float sh[8];
  __shared__ float s_scale;
  const long off = (long)blockIdx.x * N;
  const int tid = threadIdx.x;
  float ss = 0.f;
  for (int j = tid; j < N; j += 256) { float v = k[off + j]; ss += v * v; }
  ss = wred(ss);
  if ((tid & 31) == 0) sh[tid >> 5] = ss;
  __syncthreads();
  if (tid == 0) {
    float S = 0.f;
#pragma unroll
    for (int i = 0; i < 8; ++i) S += sh[i];
    float nrm = sqrtf(S);
    s_scale = 1.f / (nrm > 1e-5f ? nrm : 1e-5f);
  }
  __syncthreads();
  const float sc = s_scale;
  for (int j = tid; j < N; j += 256) k[off + j] *= sc;
}

// v_dyn = gate * (v - v_ret)   (v_ret already contains k@mem / sqrt(E))
__global__ void vdyn_kernel(const float* __restrict__ gate, const float* __restrict__ v,
                            float* __restrict__ vret, long n) {
  long i = (long)blockIdx.x * blockDim.x + threadIdx.x;
  if (i < n) vret[i] = gate[i] * (v[i] - vret[i]);
}

// ---------------------------------------------------------------------------
static inline void launch_gemm(hipStream_t st, const float* A, const float* Bm,
                               const float* bias, const float* add, float* C,
                               int M, int N, int K, int batch,
                               long sA, long sB, long sC, long sAdd,
                               int transB, int act, float alpha, float beta,
                               int cmask, int ck) {
  dim3 grid(N / BN, M / BM, batch), blk(256);
  wmma_gemm_kernel<<<grid, blk, 0, st>>>(A, Bm, bias, add, C, N, K,
                                         sA, sB, sC, sAdd, transB, act,
                                         alpha, beta, cmask, ck);
}

extern "C" void kernel_launch(void* const* d_in, const int* in_sizes, int n_in,
                              void* d_out, int out_size, void* d_ws, size_t ws_size,
                              hipStream_t stream) {
  (void)in_sizes; (void)n_in; (void)out_size; (void)ws_size;
  const int Bc = 4, Lc = 2048, Ec = 1024, Hc = 16, FFc = 2816, E4 = 4096;
  const long BL = (long)Bc * Lc;                 // 8192
  const float scale = 32.f;                      // sqrt(E)

  const float* x      = (const float*)d_in[0];
  const float* g_norm = (const float*)d_in[1];
  const float* b_norm = (const float*)d_in[2];
  const float* W_ffn1 = (const float*)d_in[3];
  const float* b_ffn1 = (const float*)d_in[4];
  const float* W_ffn2 = (const float*)d_in[5];
  const float* b_ffn2 = (const float*)d_in[6];
  const float* g_head = (const float*)d_in[7];
  const float* b_head = (const float*)d_in[8];
  const float* mem    = (const float*)d_in[9];
  const float* g_mem  = (const float*)d_in[10];
  const float* b_mem  = (const float*)d_in[11];
  const float* W_q    = (const float*)d_in[12];
  const float* b_q    = (const float*)d_in[13];
  const float* W_k    = (const float*)d_in[14];
  const float* b_k    = (const float*)d_in[15];
  const float* W_v    = (const float*)d_in[16];
  const float* b_v    = (const float*)d_in[17];
  const float* W_g    = (const float*)d_in[18];
  const float* b_g    = (const float*)d_in[19];
  // d_in[20], d_in[21]: W_mg/b_mg unused by the reference
  const float* W_o    = (const float*)d_in[22];
  const float* b_o    = (const float*)d_in[23];
  const float* W_op   = (const float*)d_in[24];
  const float* b_op   = (const float*)d_in[25];
  const float* g_ctx  = (const float*)d_in[26];
  const float* b_ctx  = (const float*)d_in[27];

  // workspace layout
  float* wsf = (float*)d_ws;
  size_t cur = 0;
  auto alloc = [&](size_t n) { float* p = wsf + cur; cur += n; return p; };
  float* xnorm   = alloc(BL * Ec);
  float* xmlp    = alloc(BL * Ec);
  float* h1      = alloc(BL * (long)FFc);
  float* xc      = alloc(BL * Ec);
  float* context = alloc(BL * Ec);
  float* ctx     = alloc(BL * Ec);
  float* q       = alloc(BL * Ec);
  float* k       = alloc(BL * Ec);
  float* v       = alloc(BL * Ec);
  float* gt      = alloc(BL * Ec);
  float* vret    = alloc(BL * Ec);   // becomes v_dyn in place
  float* mop     = alloc(BL * Ec);
  float* memo    = alloc(BL * Ec);
  float* o1      = alloc(BL * (long)E4);
  float* out2    = alloc(BL * Ec);
  float* gated   = xnorm;            // xnorm dead after FFN1

  float* y        = (float*)d_out;                  // (B,L,E)
  float* scoresO  = (float*)d_out + BL * Ec;        // (B,L,L)

  // 1) x_norm = LN(x)
  ln_kernel<<<(int)BL, 256, 0, stream>>>(x, nullptr, nullptr, g_norm, b_norm, xnorm, Ec);
  // 2) h1 = silu(x_norm @ W_ffn1 + b1)
  launch_gemm(stream, xnorm, W_ffn1, b_ffn1, nullptr, h1, (int)BL, FFc, Ec, 1,
              0, 0, 0, 0, 0, 1, 1.f, 0.f, 0, 0);
  // 3) x_mlp = x + h1 @ W_ffn2 + b2
  launch_gemm(stream, h1, W_ffn2, b_ffn2, x, xmlp, (int)BL, Ec, FFc, 1,
              0, 0, 0, 0, 0, 0, 1.f, 1.f, 0, 0);
  // 4) xc = cumsum_L(x_mlp)
  cumsum_kernel<<<(Bc * Ec + 255) / 256, 256, 0, stream>>>(xmlp, xc, Bc, Lc, Ec);
  // 5) gated = x_h * sigmoid(align)
  {
    int rows = Bc * Lc * Hc;
    align_gate_kernel<<<(rows * 32 + 255) / 256, 256, 0, stream>>>(
        xmlp, xc, g_head, b_head, gated, rows);
  }
  // 6) context = cumsum_L(gated)
  cumsum_kernel<<<(Bc * Ec + 255) / 256, 256, 0, stream>>>(gated, context, Bc, Lc, Ec);
  // 7) ctx = LN(x_mlp + context)
  ln_kernel<<<(int)BL, 256, 0, stream>>>(xmlp, context, nullptr, g_mem, b_mem, ctx, Ec);
  // 8) q,k,v,gate projections
  launch_gemm(stream, ctx, W_q, b_q, nullptr, q,  (int)BL, Ec, Ec, 1, 0,0,0,0, 0, 0, 1.f, 0.f, 0, 0);
  launch_gemm(stream, ctx, W_k, b_k, nullptr, k,  (int)BL, Ec, Ec, 1, 0,0,0,0, 0, 0, 1.f, 0.f, 0, 0);
  launch_gemm(stream, ctx, W_v, b_v, nullptr, v,  (int)BL, Ec, Ec, 1, 0,0,0,0, 0, 0, 1.f, 0.f, 0, 0);
  launch_gemm(stream, ctx, W_g, b_g, nullptr, gt, (int)BL, Ec, Ec, 1, 0,0,0,0, 0, 2, 1.f, 0.f, 0, 0);
  // 9) k row-normalize
  knorm_kernel<<<(int)BL, 256, 0, stream>>>(k, Ec);
  // 10) v_ret = (k @ mem) / scale
  launch_gemm(stream, k, mem, nullptr, nullptr, vret, (int)BL, Ec, Ec, 1,
              0, 0, 0, 0, 0, 0, 1.f / scale, 0.f, 0, 0);
  // 11) v_dyn = gate * (v - v_ret)  (in place over vret)
  vdyn_kernel<<<(int)((BL * Ec + 255) / 256), 256, 0, stream>>>(gt, v, vret, BL * Ec);
  // 12) mem_out_prev/scale = (q @ mem) / scale
  launch_gemm(stream, q, mem, nullptr, nullptr, mop, (int)BL, Ec, Ec, 1,
              0, 0, 0, 0, 0, 0, 1.f / scale, 0.f, 0, 0);
  // 13) scores = tril(q @ k^T)   (batched, output #2)
  launch_gemm(stream, q, k, nullptr, nullptr, scoresO, Lc, Lc, Ec, Bc,
              (long)Lc * Ec, (long)Lc * Ec, (long)Lc * Lc, 0,
              1, 0, 1.f, 0.f, 1, 0);
  // 14) mem_out = mop + (scores @ v_dyn) / scale^2   (K-loop truncated causally)
  launch_gemm(stream, scoresO, vret, nullptr, mop, memo, Lc, Ec, Lc, Bc,
              (long)Lc * Lc, (long)Lc * Ec, (long)Lc * Ec, (long)Lc * Ec,
              0, 0, 1.f / (scale * scale), 1.f, 0, 1);
  // 15) o1 = silu(mem_out @ W_o + b_o)
  launch_gemm(stream, memo, W_o, b_o, nullptr, o1, (int)BL, E4, Ec, 1,
              0, 0, 0, 0, 0, 1, 1.f, 0.f, 0, 0);
  // 16) out2 = o1 @ W_op + b_op
  launch_gemm(stream, o1, W_op, b_op, nullptr, out2, (int)BL, Ec, E4, 1,
              0, 0, 0, 0, 0, 0, 1.f, 0.f, 0, 0);
  // 17) y = x_mlp + LN(out2)
  ln_kernel<<<(int)BL, 256, 0, stream>>>(out2, nullptr, xmlp, g_ctx, b_ctx, y, Ec);
}
